// DoubleLSTM_32530082300458
// MI455X (gfx1250) — matrix operand order repaired and checked by reading, not compiled, not run on hardware
//
#include <hip/hip_runtime.h>
#include <hip/hip_bf16.h>

typedef __attribute__((ext_vector_type(16))) _Float16 v16h;
typedef __attribute__((ext_vector_type(8)))  float    v8f;

#define B_  256
#define T_  1024
#define E_  16
#define H_  24
#define C_  128
#define G4H 96    // 4*H
#define D2H 48    // 2*H
#define F6H 144   // 6*H
#define HP  32    // padded H stride (f16) in LDS
#define MP  160   // padded 6H stride (f16) in LDS

// ---- CDNA5 WMMA 16x16x32 f16 fragment maps (ISA 7.12.2), simplified ----
// A 16x32 f16: lane row m = lane&15; element j -> k = (j>=8?16:0) + 8*(lane>=16) + (j&7)
// B 32x16 f16: lane col n = lane&15; element j -> k = (j>=8?16:0) + 8*(lane>=16) + (j&7)
// C/D 16x16 f32: element r -> m = 8*(lane>=16)+r, n = lane&15

__device__ __forceinline__ float sigf(float x) {
  return __builtin_amdgcn_rcpf(1.0f + __expf(-x));
}
__device__ __forceinline__ float tanh_fast(float x) {
  return 1.0f - 2.0f * __builtin_amdgcn_rcpf(1.0f + __expf(2.0f * x));
}

__device__ __forceinline__ v8f wmma_f16(v16h a, v16h b, v8f c) {
  return __builtin_amdgcn_wmma_f32_16x16x32_f16(false, a, false, b, (short)0, c,
                                                false, false);
}

// A fragment from an f16 row (global or LDS): fully-valid 32-wide K chunk.
__device__ __forceinline__ v16h a_frag_h16(const _Float16* __restrict__ p, int half) {
  v16h a;
#pragma unroll
  for (int j = 0; j < 16; ++j)
    a[j] = p[(j >= 8 ? 16 : 0) + (half << 3) + (j & 7)];
  return a;
}
// A fragment from an f16 row: only K [0,16) valid (upper half zero).
__device__ __forceinline__ v16h a_frag_h16_lo(const _Float16* __restrict__ p, int half) {
  v16h a;
#pragma unroll
  for (int j = 0; j < 8; ++j) a[j] = p[(half << 3) + j];
#pragma unroll
  for (int j = 8; j < 16; ++j) a[j] = (_Float16)0.f;
  return a;
}
// Convert two prefetched float4s (8 f32) into a lo-half A fragment.
__device__ __forceinline__ v16h a_frag_cvt_lo(float4 x0, float4 x1) {
  v16h a;
  a[0] = (_Float16)x0.x; a[1] = (_Float16)x0.y;
  a[2] = (_Float16)x0.z; a[3] = (_Float16)x0.w;
  a[4] = (_Float16)x1.x; a[5] = (_Float16)x1.y;
  a[6] = (_Float16)x1.z; a[7] = (_Float16)x1.w;
#pragma unroll
  for (int j = 8; j < 16; ++j) a[j] = (_Float16)0.f;
  return a;
}

// ---- Weight pre-pack: fragment-ordered f16, zero-padded ----
__global__ void pack_b_kernel(const float* __restrict__ W, _Float16* __restrict__ dst,
                              int LEN, int ntiles, int nchunks) {
  int idx = blockIdx.x * blockDim.x + threadIdx.x;
  int total = ntiles * nchunks * 32;
  if (idx >= total) return;
  int lane = idx & 31;
  int tc = idx >> 5;
  int c = tc % nchunks;
  int ncol = (tc / nchunks) * 16 + (lane & 15);
  int kbase = c * 32 + ((lane >> 4) << 3);
#pragma unroll
  for (int j = 0; j < 16; ++j) {
    int k = kbase + (j >= 8 ? 16 : 0) + (j & 7);
    dst[idx * 16 + j] = (k < LEN) ? (_Float16)W[ncol * LEN + k] : (_Float16)0.f;
  }
}

// ---------- Fused LSTM scan, both directions of one layer ----------
// gridDim = (16, 2): x = 16-row batch tile, y = direction (0 fwd, 1 rev).
// in: [B,T,IN] (f32 layer0 / f16 layer1); h0/c0: [2,B,24] f32 for this layer.
// out: [B,T,48] f16; dir d writes cols [24d, 24d+24) at original time order.
template <int IN>
__global__ __launch_bounds__(192) void lstm_scan2_kernel(
    const void* __restrict__ in_v,
    const _Float16* __restrict__ wihpF, const _Float16* __restrict__ whhpF,
    const _Float16* __restrict__ wihpR, const _Float16* __restrict__ whhpR,
    const float* __restrict__ bihF, const float* __restrict__ bhhF,
    const float* __restrict__ bihR, const float* __restrict__ bhhR,
    const float* __restrict__ h0, const float* __restrict__ c0,
    _Float16* __restrict__ out) {
  constexpr int NC = (IN + 31) / 32;
  __shared__ _Float16 h_sh[16 * HP];  // hidden state f16, zero-padded cols 24..31
  __shared__ float g_sh[16 * G4H];    // gate outputs [16][96]

  const int dir  = blockIdx.y;
  const int rev  = dir;
  const int ooff = dir * H_;
  const _Float16* wihp = dir ? wihpR : wihpF;
  const _Float16* whhp = dir ? whhpR : whhpF;
  const float* bih = dir ? bihR : bihF;
  const float* bhh = dir ? bhhR : bhhF;
  const float* h0d = h0 + (size_t)dir * B_ * H_;
  const float* c0d = c0 + (size_t)dir * B_ * H_;

  const int tid  = threadIdx.x;
  const int lane = tid & 31;
  const int wave = tid >> 5;  // 0..5
  const int half = lane >> 4;
  const int mrow = lane & 15;
  const int b0   = blockIdx.x * 16;
  const int ncol = wave * 16 + mrow;

  for (int e = tid; e < 16 * HP; e += 192) {
    int m = e / HP, j = e % HP;
    h_sh[e] = (j < H_) ? (_Float16)h0d[(b0 + m) * H_ + j] : (_Float16)0.f;
  }
  float creg[2];
#pragma unroll
  for (int s = 0; s < 2; ++s) {
    int e = tid * 2 + s;
    creg[s] = c0d[(b0 + e / H_) * H_ + e % H_];
  }

  // packed weight fragments: single aligned 32B loads
  v16h bwih[NC];
#pragma unroll
  for (int c = 0; c < NC; ++c)
    bwih[c] = *(const v16h*)(wihp + ((size_t)(wave * NC + c) * 32 + lane) * 16);
  const v16h bwhh = *(const v16h*)(whhp + ((size_t)wave * 32 + lane) * 16);
  v8f cbias;
  {
    float bv = bih[ncol] + bhh[ncol];
#pragma unroll
    for (int r = 0; r < 8; ++r) cbias[r] = bv;
  }
  __syncthreads();

  // software-pipelined input fragments (x loads are h-independent)
  float4 px0, px1;    // layer 0 raw prefetch
  v16h pa0, pa1;      // layer 1 prefetch (fragment == raw load)
  {
    const int tt0 = rev ? (T_ - 1) : 0;
    if constexpr (IN == 16) {
      const float* p = (const float*)in_v +
                       ((size_t)(b0 + mrow) * T_ + tt0) * IN + (half << 3);
      px0 = *(const float4*)p;
      px1 = *(const float4*)(p + 4);
    } else {
      const _Float16* p =
          (const _Float16*)in_v + ((size_t)(b0 + mrow) * T_ + tt0) * IN;
      pa0 = a_frag_h16(p, half);
      pa1 = a_frag_h16_lo(p + 32, half);
    }
  }

  for (int t = 0; t < T_; ++t) {
    const int tt = rev ? (T_ - 1 - t) : t;
    v8f acc = cbias;

    // input projection with prefetched data
    if constexpr (IN == 16) {
      acc = wmma_f16(a_frag_cvt_lo(px0, px1), bwih[0], acc);
    } else {
      acc = wmma_f16(pa0, bwih[0], acc);
      acc = wmma_f16(pa1, bwih[1], acc);
    }

    // issue next step's input loads now; latency hides behind the rest of
    // this step (h-WMMA, gate exchange, 2 barriers, cell update)
    {
      const int tn  = (t + 1 < T_) ? (t + 1) : t;
      const int ttn = rev ? (T_ - 1 - tn) : tn;
      if constexpr (IN == 16) {
        const float* p = (const float*)in_v +
                         ((size_t)(b0 + mrow) * T_ + ttn) * IN + (half << 3);
        px0 = *(const float4*)p;
        px1 = *(const float4*)(p + 4);
      } else {
        const _Float16* p =
            (const _Float16*)in_v + ((size_t)(b0 + mrow) * T_ + ttn) * IN;
        pa0 = a_frag_h16(p, half);
        pa1 = a_frag_h16_lo(p + 32, half);
      }
    }

    // recurrent projection (padded f16 LDS, pure ds loads)
    acc = wmma_f16(a_frag_h16(&h_sh[mrow * HP], half), bwhh, acc);

#pragma unroll
    for (int r = 0; r < 8; ++r) g_sh[(half * 8 + r) * G4H + ncol] = acc[r];
    __syncthreads();

    // LSTM cell update (torch gate order i,f,g,o)
#pragma unroll
    for (int s = 0; s < 2; ++s) {
      int e = tid * 2 + s;
      int m = e / H_, j = e % H_;
      float gi = g_sh[m * G4H + j];
      float gf = g_sh[m * G4H + H_ + j];
      float gg = g_sh[m * G4H + 2 * H_ + j];
      float go = g_sh[m * G4H + 3 * H_ + j];
      float cn = sigf(gf) * creg[s] + sigf(gi) * tanh_fast(gg);
      creg[s] = cn;
      float hn = sigf(go) * tanh_fast(cn);
      h_sh[m * HP + j] = (_Float16)hn;
      out[((size_t)(b0 + m) * T_ + tt) * D2H + ooff + j] = (_Float16)hn;
    }
    __syncthreads();
  }
}

// ---------- Fused head: relu(relu(o1@W1^T+b1)@W2^T+b2) ----------
__global__ __launch_bounds__(256) void head_kernel(
    const _Float16* __restrict__ o1,
    const _Float16* __restrict__ w1p, const float* __restrict__ b1,
    const _Float16* __restrict__ w2p, const float* __restrict__ b2,
    float* __restrict__ out) {
  __shared__ _Float16 hmid_sh[16 * MP];  // [16][160], cols 144..159 zero
  const int tid  = threadIdx.x;
  const int lane = tid & 31;
  const int wave = tid >> 5;  // 0..7
  const int half = lane >> 4;
  const int mrow = lane & 15;
  const int n0   = blockIdx.x * 16;

  hmid_sh[(tid >> 4) * MP + F6H + (tid & 15)] = (_Float16)0.f;

  const _Float16* arow = o1 + (size_t)(n0 + mrow) * D2H;
  const v16h a0 = a_frag_h16(arow, half);
  const v16h a1 = a_frag_h16_lo(arow + 32, half);

  for (int tile = wave; tile < 9; tile += 8) {
    const int ncol = tile * 16 + mrow;
    v8f acc;
    float bv = b1[ncol];
#pragma unroll
    for (int r = 0; r < 8; ++r) acc[r] = bv;
    acc = wmma_f16(a0, *(const v16h*)(w1p + ((size_t)(tile * 2 + 0) * 32 + lane) * 16), acc);
    acc = wmma_f16(a1, *(const v16h*)(w1p + ((size_t)(tile * 2 + 1) * 32 + lane) * 16), acc);
#pragma unroll
    for (int r = 0; r < 8; ++r)
      hmid_sh[(half * 8 + r) * MP + ncol] = (_Float16)fmaxf(acc[r], 0.f);
  }
  __syncthreads();

  {
    const int ncol = wave * 16 + mrow;
    v8f acc;
    float bv = b2[ncol];
#pragma unroll
    for (int r = 0; r < 8; ++r) acc[r] = bv;
#pragma unroll
    for (int c = 0; c < 5; ++c) {
      v16h a = a_frag_h16(&hmid_sh[mrow * MP + c * 32], half);
      v16h b = *(const v16h*)(w2p + ((size_t)(wave * 5 + c) * 32 + lane) * 16);
      acc = wmma_f16(a, b, acc);
    }
#pragma unroll
    for (int r = 0; r < 8; ++r)
      out[(size_t)(n0 + half * 8 + r) * C_ + ncol] = fmaxf(acc[r], 0.f);
  }
}

extern "C" void kernel_launch(void* const* d_in, const int* in_sizes, int n_in,
                              void* d_out, int out_size, void* d_ws, size_t ws_size,
                              hipStream_t stream) {
  const float* x  = (const float*)d_in[0];
  const float* h0 = (const float*)d_in[1];  // [4,B,H]
  const float* c0 = (const float*)d_in[2];
  const float *Wih[4], *Whh[4], *bih[4], *bhh[4];
  for (int d = 0; d < 4; ++d) {
    Wih[d] = (const float*)d_in[3 + 4 * d + 0];
    Whh[d] = (const float*)d_in[3 + 4 * d + 1];
    bih[d] = (const float*)d_in[3 + 4 * d + 2];
    bhh[d] = (const float*)d_in[3 + 4 * d + 3];
  }
  const float* W1 = (const float*)d_in[19];
  const float* b1 = (const float*)d_in[20];
  const float* W2 = (const float*)d_in[21];
  const float* b2 = (const float*)d_in[22];
  float* out = (float*)d_out;

  // workspace (all f16), every offset 32B-aligned
  const size_t OSZ = (size_t)B_ * T_ * D2H;
  _Float16* o0  = (_Float16*)d_ws;
  _Float16* o1  = o0 + OSZ;
  _Float16* w1p = o1 + OSZ;
  _Float16* w2p = w1p + 9 * 2 * 512;
  _Float16* wp  = w2p + 8 * 5 * 512;
  _Float16 *wihp[4], *whhp[4];
  for (int d = 0; d < 4; ++d) {
    wihp[d] = wp + d * (6 * 2 * 512 + 6 * 512);
    whhp[d] = wihp[d] + 6 * 2 * 512;
  }

  auto pack = [&](const float* W, _Float16* dst, int LEN, int ntiles, int nchunks) {
    int total = ntiles * nchunks * 32;
    pack_b_kernel<<<(total + 255) / 256, 256, 0, stream>>>(W, dst, LEN, ntiles, nchunks);
  };
  pack(W1, w1p, D2H, 9, 2);
  pack(W2, w2p, F6H, 8, 5);
  for (int d = 0; d < 2; ++d) pack(Wih[d], wihp[d], E_, 6, 1);
  for (int d = 2; d < 4; ++d) pack(Wih[d], wihp[d], D2H, 6, 2);
  for (int d = 0; d < 4; ++d) pack(Whh[d], whhp[d], H_, 6, 1);

  const size_t BH = (size_t)B_ * H_;
  dim3 sgrid(B_ / 16, 2), sblock(192);

  // layer 0: both directions concurrently (grid.y = dir)
  lstm_scan2_kernel<16><<<sgrid, sblock, 0, stream>>>(
      x, wihp[0], whhp[0], wihp[1], whhp[1],
      bih[0], bhh[0], bih[1], bhh[1],
      h0 + 0 * BH, c0 + 0 * BH, o0);
  // layer 1: both directions concurrently
  lstm_scan2_kernel<48><<<sgrid, sblock, 0, stream>>>(
      o0, wihp[2], whhp[2], wihp[3], whhp[3],
      bih[2], bhh[2], bih[3], bhh[3],
      h0 + 2 * BH, c0 + 2 * BH, o1);
  head_kernel<<<dim3((B_ * T_) / 16), 256, 0, stream>>>(o1, w1p, b1, w2p, b2, out);
}
